// Transformer_71897752535052
// MI455X (gfx1250) — matrix operand order repaired and checked by reading, not compiled, not run on hardware
//
// MI455X / gfx1250 transformer forward. All matmuls via v_wmma_f32_16x16x32_f16
// (f16 inputs, f32 accum). Inner loops are branch-free b128 loads. All GEMM
// B-operands (packed weights, K-proj, V^T) are stored f16: halves the dominant
// L2 traffic term (B re-fetched per M-tile) and removes half the cvt work.
#include <hip/hip_runtime.h>

typedef __attribute__((ext_vector_type(16))) _Float16 v16h;
typedef __attribute__((ext_vector_type(8)))  _Float16 v8h;
typedef __attribute__((ext_vector_type(8)))  float    v8f;

#define TE    256            // embed
#define THN   8              // heads
#define THD   32             // head dim
#define TFF   1024           // ffn hidden
#define TLAY  4              // layers
#define TVOUT 128
#define TSEQ  48
#define TBAT  512
#define TMROW (TBAT * TSEQ)  // 24576 token rows

// ---------------------------------------------------------------------------
// Generic batched GEMM. A is f32 K-contiguous, B is f16 K-contiguous:
//   A addr: offA(z) + m*lda + k
//   B addr: offB(z) + n*ldb + k                      (halves)
//   C addr: offC(z) + m*ldc + (n&cmask)*scn + (n>>csh)*scb   (+bias,+R,relu)
//   off(z) = (z>>zsh)*zo + (z&zmk)*zi
// C stored f32 (C) or f16 (C16). One wave computes a 16x(16*NT) strip:
// one A fragment feeds NT WMMAs.
// ---------------------------------------------------------------------------
struct GemmP {
  const float*     A;
  const _Float16*  Bm;
  const float*     bias;   // length N (or null)
  const float*     R;      // residual, [M,N] row-major, ldr (or null)
  float*           C;      // f32 output (or null)
  _Float16*        C16;    // f16 output (or null)
  int M, N, K;
  int lda, ldb, ldc, ldr;
  int cmask, scn, csh, scb;
  int zshA, zmkA; long long zoA, ziA;
  int zshB, zmkB; long long zoB, ziB;
  int zshC, zmkC; long long zoC, ziC;
  int relu;
};

__device__ __forceinline__ float4 ldg4(const float* p) {
  return *(const float4*)p;
}
// Guarded f32x4 load: unconditional load from clamped address + value select.
// Valid for K multiple of 4 (all our shapes): per-float4 all-or-nothing.
__device__ __forceinline__ float4 ldg4z(const float* base, int k, int K) {
  const bool ok = (k + 4) <= K;
  const float4 t = *(const float4*)(base + (ok ? k : 0));
  float4 r;
  r.x = ok ? t.x : 0.f; r.y = ok ? t.y : 0.f;
  r.z = ok ? t.z : 0.f; r.w = ok ? t.w : 0.f;
  return r;
}
// Guarded f16x8 load (one b128): K multiple of 8 -> per-chunk all-or-nothing.
__device__ __forceinline__ v8h ldh8z(const _Float16* base, int k, int K) {
  const bool ok = (k + 8) <= K;
  const v8h t = *(const v8h*)(base + (ok ? k : 0));
  const v8h z = {};
  return ok ? t : z;
}
__device__ __forceinline__ v16h cvt16(float4 a, float4 b, float4 c, float4 d) {
  v16h r;
  r[0]  = (_Float16)a.x; r[1]  = (_Float16)a.y; r[2]  = (_Float16)a.z; r[3]  = (_Float16)a.w;
  r[4]  = (_Float16)b.x; r[5]  = (_Float16)b.y; r[6]  = (_Float16)b.z; r[7]  = (_Float16)b.w;
  r[8]  = (_Float16)c.x; r[9]  = (_Float16)c.y; r[10] = (_Float16)c.z; r[11] = (_Float16)c.w;
  r[12] = (_Float16)d.x; r[13] = (_Float16)d.y; r[14] = (_Float16)d.z; r[15] = (_Float16)d.w;
  return r;
}
__device__ __forceinline__ v16h cat16(v8h lo, v8h hi) {
  return __builtin_shufflevector(lo, hi, 0, 1, 2, 3, 4, 5, 6, 7,
                                 8, 9, 10, 11, 12, 13, 14, 15);
}

template <int NT>
__global__ __launch_bounds__(256) void gemm_f16_wmma(GemmP p) {
  const int tn  = p.N >> 4;
  const int tnw = tn / NT;
  const int ntiles = (p.M >> 4) * tnw;
  const int wave = (int)(blockIdx.x * 8u + (threadIdx.x >> 5));
  if (wave >= ntiles) return;                // wave-uniform: EXEC stays all-1s
  const int lane = (int)(threadIdx.x & 31u);
  const int tileM  = (wave / tnw) << 4;
  const int tileN0 = (wave % tnw) * (16 * NT);
  const long long z = (long long)blockIdx.z;

  const long long offA = (z >> p.zshA) * p.zoA + (long long)(z & p.zmkA) * p.ziA;
  const long long offB = (z >> p.zshB) * p.zoB + (long long)(z & p.zmkB) * p.ziB;
  const long long offC = (z >> p.zshC) * p.zoC + (long long)(z & p.zmkC) * p.ziC;

  const int loc = lane & 15;   // M row (A) / N col (B,C) within 16-tile
  const int hk  = lane >> 4;   // lane-half selects K sub-range

  const float* __restrict__ Arow = p.A + offA + (long long)(tileM + loc) * p.lda;
  const _Float16* Brow[NT];
#pragma unroll
  for (int j = 0; j < NT; ++j)
    Brow[j] = p.Bm + offB + (long long)(tileN0 + j * 16 + loc) * p.ldb;

  v8f acc[NT];
#pragma unroll
  for (int j = 0; j < NT; ++j) {
    v8f zf = {0.f, 0.f, 0.f, 0.f, 0.f, 0.f, 0.f, 0.f};
    acc[j] = zf;
  }

  // A f16 16x32 layout (matches B): lanes0-15 K 0..15, lanes16-31 K 16..31
  const int kA0 = hk * 16;
  const int kA1 = hk * 16 + 8;
  // B f16 32x16 layout: lanes0-15 K 0..15, lanes16-31 K 16..31
  const int kB0 = hk * 16;

  const int kmain = p.K & ~31;
  for (int kk = 0; kk < kmain; kk += 32) {
    const float* ap = Arow + kk;
    const v16h af = cvt16(ldg4(ap + kA0), ldg4(ap + kA0 + 4),
                          ldg4(ap + kA1), ldg4(ap + kA1 + 4));
#pragma unroll
    for (int j = 0; j < NT; ++j) {
      const _Float16* bp = Brow[j] + kk + kB0;
      const v16h bf = cat16(*(const v8h*)bp, *(const v8h*)(bp + 8));
      acc[j] = __builtin_amdgcn_wmma_f32_16x16x32_f16(false, af, false, bf,
                                                      (short)0, acc[j], false, false);
    }
  }
  if (kmain < p.K) {           // K=48 tail: clamped loads + cndmask, no branches
    const v16h af = cvt16(ldg4z(Arow, kmain + kA0,     p.K),
                          ldg4z(Arow, kmain + kA0 + 4, p.K),
                          ldg4z(Arow, kmain + kA1,     p.K),
                          ldg4z(Arow, kmain + kA1 + 4, p.K));
#pragma unroll
    for (int j = 0; j < NT; ++j) {
      const int kb = kmain + kB0;
      const v16h bf = cat16(ldh8z(Brow[j], kb, p.K), ldh8z(Brow[j], kb + 8, p.K));
      acc[j] = __builtin_amdgcn_wmma_f32_16x16x32_f16(false, af, false, bf,
                                                      (short)0, acc[j], false, false);
    }
  }

#pragma unroll
  for (int j = 0; j < NT; ++j) {
    const int nn = tileN0 + j * 16 + loc;
    const long long ccol = offC + (long long)(nn & p.cmask) * p.scn
                                + (long long)((unsigned)nn >> p.csh) * p.scb;
    const float bv = p.bias ? p.bias[nn] : 0.f;
#pragma unroll
    for (int v = 0; v < 8; ++v) {
      const int mrow = tileM + v + (hk << 3);   // D layout: VGPR v -> M = v / v+8
      float val = acc[j][v] + bv;
      if (p.R)    val += p.R[(long long)mrow * p.ldr + nn];
      if (p.relu) val  = fmaxf(val, 0.f);
      const long long ci = ccol + (long long)mrow * p.ldc;
      if (p.C16) p.C16[ci] = (_Float16)val;
      else       p.C[ci]   = val;
    }
  }
}

// ---------------------------------------------------------------------------
// Weight pack (f32 -> f16): Wt[n*K + k] = W[(n>>nsh)*sb + (n&nmask)*sn + k*sk]
// Makes every GEMM B operand K-contiguous f16 (b128-loadable, half traffic).
// ---------------------------------------------------------------------------
__global__ void pack_w(const float* __restrict__ W, _Float16* __restrict__ Wt,
                       int N, int K, int sk, int sn, int sb, int nsh, int nmask) {
  const long long i = (long long)blockIdx.x * blockDim.x + threadIdx.x;
  if (i >= (long long)N * K) return;
  const int k = (int)(i % K);
  const int n = (int)(i / K);
  Wt[i] = (_Float16)W[(long long)((unsigned)n >> nsh) * sb
                      + (long long)(n & nmask) * sn + (long long)k * sk];
}

// ---------------------------------------------------------------------------
// LayerNorm over E=256: one wave per row, 8 elems/lane, shfl_xor reduction.
// ---------------------------------------------------------------------------
__global__ __launch_bounds__(256) void layernorm_k(const float* __restrict__ x,
                                                   const float* __restrict__ g,
                                                   const float* __restrict__ b,
                                                   float* __restrict__ y, int rows) {
  const int row = (int)(blockIdx.x * 8u + (threadIdx.x >> 5));
  if (row >= rows) return;
  const int lane = (int)(threadIdx.x & 31u);
  const float* xr = x + (long long)row * TE;
  float v[8];
  float s = 0.f;
#pragma unroll
  for (int i = 0; i < 8; ++i) { v[i] = xr[lane + 32 * i]; s += v[i]; }
#pragma unroll
  for (int o = 16; o > 0; o >>= 1) s += __shfl_xor(s, o, 32);
  const float mean = s * (1.f / TE);
  float q = 0.f;
#pragma unroll
  for (int i = 0; i < 8; ++i) { const float d = v[i] - mean; q += d * d; }
#pragma unroll
  for (int o = 16; o > 0; o >>= 1) q += __shfl_xor(q, o, 32);
  const float inv = rsqrtf(q * (1.f / TE) + 1e-5f);
  float* yr = y + (long long)row * TE;
#pragma unroll
  for (int i = 0; i < 8; ++i) {
    const int e = lane + 32 * i;
    yr[e] = (v[i] - mean) * inv * g[e] + b[e];
  }
}

// ---------------------------------------------------------------------------
// Attention softmax: rows = B*H*T of length T. scale = E^-0.5 = 1/16 (reference
// scales by embed_size, not head_dim). causal -> tril mask (s<=t), masked -> 0.
// ---------------------------------------------------------------------------
__global__ void softmax_wei(float* __restrict__ w, int rows, int causal) {
  const int r = (int)(blockIdx.x * blockDim.x + threadIdx.x);
  if (r >= rows) return;
  float* wr = w + (long long)r * TSEQ;
  const int t = r % TSEQ;
  const int lim = causal ? (t + 1) : TSEQ;
  const float scale = 0.0625f;
  float mx = -3.0e38f;
  for (int s = 0; s < lim; ++s) mx = fmaxf(mx, wr[s] * scale);
  float sum = 0.f;
  for (int s = 0; s < lim; ++s) { const float e = __expf(wr[s] * scale - mx); wr[s] = e; sum += e; }
  const float inv = 1.f / sum;
  for (int s = 0; s < lim; ++s) wr[s] *= inv;
  for (int s = lim; s < TSEQ; ++s) wr[s] = 0.f;
}

// ---------------------------------------------------------------------------
// Embedding lookup * sqrt(E) + sinusoidal positional encoding.
// ---------------------------------------------------------------------------
__global__ void embed_pe(const int* __restrict__ tok, const float* __restrict__ emb,
                         float* __restrict__ out) {
  const long long i = (long long)blockIdx.x * blockDim.x + threadIdx.x;
  if (i >= (long long)TMROW * TE) return;
  const int e = (int)(i & (TE - 1));
  const long long m = i >> 8;
  const int t = (int)(m % TSEQ);
  float v = emb[(long long)tok[m] * TE + e] * 16.0f;   // sqrt(256)
  const float ang = (float)t * __expf((float)(e & ~1) * (-9.210340371976184f / 256.0f));
  v += (e & 1) ? cosf(ang) : sinf(ang);
  out[i] = v;
}

// ---------------------------------------------------------------------------
// Final softmax over the SEQUENCE axis: for each (b, vocab) normalize across t.
// ---------------------------------------------------------------------------
__global__ void softmax_seq(const float* __restrict__ logits, float* __restrict__ out) {
  const int i = (int)(blockIdx.x * blockDim.x + threadIdx.x);
  if (i >= TBAT * TVOUT) return;
  const int b = i / TVOUT, v = i % TVOUT;
  const float* base = logits + (long long)b * TSEQ * TVOUT + v;
  float mx = -3.0e38f;
  for (int t = 0; t < TSEQ; ++t) mx = fmaxf(mx, base[t * TVOUT]);
  float sum = 0.f;
  for (int t = 0; t < TSEQ; ++t) sum += __expf(base[t * TVOUT] - mx);
  const float inv = 1.f / sum;
  float* ob = out + (long long)b * TSEQ * TVOUT + v;
  for (int t = 0; t < TSEQ; ++t) ob[t * TVOUT] = __expf(base[t * TVOUT] - mx) * inv;
}

// ---------------------------------------------------------------------------
// Host orchestration
// ---------------------------------------------------------------------------
static GemmP gp0() {
  GemmP p{};
  p.cmask = 0x7fffffff; p.scn = 1; p.csh = 31; p.scb = 0;
  p.zshA = p.zshB = p.zshC = 31;
  p.zmkA = p.zmkB = p.zmkC = 0x7fffffff;
  p.zoA = p.zoB = p.zoC = 0; p.ziA = p.ziB = p.ziC = 0;
  return p;
}

static void launch_gemm(hipStream_t s, const GemmP& p, int batches) {
  const int tn = p.N >> 4;
  const int nt = (tn % 4 == 0) ? 4 : (tn % 3 == 0) ? 3 : (tn % 2 == 0) ? 2 : 1;
  const int ntiles = (p.M >> 4) * (tn / nt);
  dim3 grid((unsigned)((ntiles + 7) / 8), 1, (unsigned)batches);
  switch (nt) {
    case 4: gemm_f16_wmma<4><<<grid, dim3(256), 0, s>>>(p); break;
    case 3: gemm_f16_wmma<3><<<grid, dim3(256), 0, s>>>(p); break;
    case 2: gemm_f16_wmma<2><<<grid, dim3(256), 0, s>>>(p); break;
    default: gemm_f16_wmma<1><<<grid, dim3(256), 0, s>>>(p); break;
  }
}

extern "C" void kernel_launch(void* const* d_in, const int* in_sizes, int n_in,
                              void* d_out, int out_size, void* d_ws, size_t ws_size,
                              hipStream_t stream) {
  (void)in_sizes; (void)n_in; (void)out_size; (void)ws_size;
  // Inputs flattened in setup_inputs() dict insertion order, depth-first:
  //  0:src 1:trg 2:enc_emb 3:dec_emb
  //  4-8:  enc_sa {wq,wk,wv,wo,bo}   9-12: enc_ff {w1,b1,w2,b2}  13-14: enc_ln {g,b}
  // 15-19: dec_sa {wq,wk,wv,wo,bo}  20-21: dec_ln {g,b}
  // 22-26: dec_blk_sa {wq,wk,wv,wo,bo} 27-30: dec_blk_ff {w1,b1,w2,b2} 31-32: dec_blk_ln {g,b}
  // 33: out_w 34: out_b
  const int* src = (const int*)d_in[0];
  const int* trg = (const int*)d_in[1];
  auto F = [&](int i) { return (const float*)d_in[i]; };
  struct MHAW { const float *wq, *wk, *wv, *wo, *bo; };
  struct FFNW { const float *w1, *b1, *w2, *b2; };
  struct LNW  { const float *g, *b; };
  MHAW enc_sa  {F(4),  F(5),  F(6),  F(7),  F(8)};
  FFNW enc_ff  {F(9),  F(10), F(11), F(12)};
  LNW  enc_ln  {F(13), F(14)};
  MHAW dec_sa  {F(15), F(16), F(17), F(18), F(19)};
  LNW  dec_ln  {F(20), F(21)};
  MHAW dblk_sa {F(22), F(23), F(24), F(25), F(26)};
  FFNW dblk_ff {F(27), F(28), F(29), F(30)};
  LNW  dblk_ln {F(31), F(32)};
  const float* out_w = F(33);
  const float* out_b = F(34);

  // Workspace layout: f32 region then f16 region (all chunks >=64B aligned)
  float* ws = (float*)d_ws;
  size_t off = 0;
  auto take = [&](size_t n) { float* p = ws + off; off += n; return p; };
  float* encx   = take((size_t)TMROW * TE);
  float* decx   = take((size_t)TMROW * TE);
  float* qb     = take((size_t)TMROW * TE);            // [B,T,H,HD] f32 (A operand)
  float* ctx    = take((size_t)TMROW * TE);            // attn out, [B,T,H,HD]
  float* t1     = take((size_t)TMROW * TE);
  float* t2     = take((size_t)TMROW * TE);
  float* ffh    = take((size_t)TMROW * TFF);
  float* wei    = take((size_t)TBAT * THN * TSEQ * TSEQ);
  float* logits = take((size_t)TMROW * TVOUT);
  _Float16* hbase = (_Float16*)(ws + off);
  size_t hoff = 0;
  auto takeh = [&](size_t n) { _Float16* p = hbase + hoff; hoff += n; return p; };
  _Float16* kb16   = takeh((size_t)TMROW * TE);        // [B,T,H,HD] f16 (B operand)
  _Float16* vt16   = takeh((size_t)TMROW * TE);        // [B,H,HD,T] f16 (B operand)
  _Float16* wp_q   = takeh((size_t)TE * TE);           // packed weights [N][K] f16
  _Float16* wp_k   = takeh((size_t)TE * TE);
  _Float16* wp_v   = takeh((size_t)TE * TE);
  _Float16* wp_o   = takeh((size_t)TE * TE);
  _Float16* wp_f1  = takeh((size_t)TE * TFF);
  _Float16* wp_f2  = takeh((size_t)TE * TFF);
  _Float16* wp_out = takeh((size_t)TE * TVOUT);

  auto packw = [&](const float* W, _Float16* Wt, int N, int K,
                   int sk, int sn, int sb, int nsh, int nmask) {
    const long long tot = (long long)N * K;
    pack_w<<<dim3((unsigned)((tot + 255) / 256)), dim3(256), 0, stream>>>(
        W, Wt, N, K, sk, sn, sb, nsh, nmask);
  };
  // (L,H,E,HD) head-blocked -> Wt[n=h*32+d][k=e]
  auto pack_hb = [&](const float* W, int l, _Float16* Wt) {
    packw(W + (size_t)l * THN * TE * THD, Wt, TE, TE, THD, 1, TE * THD, 5, 31);
  };
  // row-major [K][N] -> Wt[n][k]
  auto pack_rm = [&](const float* W, long long loff, _Float16* Wt, int N, int K) {
    packw(W + loff, Wt, N, K, N, 1, 0, 31, 0x7fffffff);
  };

  auto layernorm = [&](const float* x, const LNW& ln, int l, float* y) {
    layernorm_k<<<dim3(TMROW / 8), dim3(256), 0, stream>>>(
        x, ln.g + (size_t)l * TE, ln.b + (size_t)l * TE, y, TMROW);
  };

  // full MHA: out = resid + (attn(qin,kin,vin) @ wo + bo)
  auto mha = [&](const float* qin, const float* kin, const float* vin,
                 const MHAW& w, int l, int causal, float* out, const float* resid) {
    pack_hb(w.wq, l, wp_q);
    pack_hb(w.wk, l, wp_k);
    pack_hb(w.wv, l, wp_v);
    pack_rm(w.wo, (size_t)l * TE * TE, wp_o, TE, TE);
    // Q projection -> qb f32 [B,T,H,HD] (consumed as A by scores)
    {
      GemmP p = gp0();
      p.A = qin; p.lda = TE; p.Bm = wp_q; p.ldb = TE;
      p.C = qb; p.ldc = TE;
      p.M = TMROW; p.N = TE; p.K = TE;
      launch_gemm(stream, p, 1);
    }
    // K projection -> kb16 f16 [B,T,H,HD] (consumed as B by scores)
    {
      GemmP p = gp0();
      p.A = kin; p.lda = TE; p.Bm = wp_k; p.ldb = TE;
      p.C16 = kb16; p.ldc = TE;
      p.M = TMROW; p.N = TE; p.K = TE;
      launch_gemm(stream, p, 1);
    }
    // V projection, batched over b, stored transposed f16 -> vt16[b][h][d][t]
    {
      GemmP p = gp0();
      p.A = vin; p.lda = TE; p.ziA = (long long)TSEQ * TE;
      p.Bm = wp_v; p.ldb = TE;
      p.C16 = vt16; p.ziC = (long long)TSEQ * TE;
      p.ldc = 1; p.cmask = 31; p.scn = TSEQ; p.csh = 5; p.scb = THD * TSEQ;
      p.M = TSEQ; p.N = TE; p.K = TE;
      launch_gemm(stream, p, TBAT);
    }
    // scores: wei[z=b*8+h] = Q_bh (48x32) @ K_bh^T (32x48)
    {
      GemmP p = gp0();
      p.A = qb; p.lda = TE;
      p.zshA = 3; p.zmkA = 7; p.zoA = (long long)TSEQ * TE; p.ziA = THD;
      p.Bm = kb16; p.ldb = TE;                    // B(k=d, n=s) = kb[..+ s*256 + d]
      p.zshB = 3; p.zmkB = 7; p.zoB = (long long)TSEQ * TE; p.ziB = THD;
      p.C = wei; p.ldc = TSEQ; p.ziC = (long long)TSEQ * TSEQ;
      p.M = TSEQ; p.N = TSEQ; p.K = THD;
      launch_gemm(stream, p, TBAT * THN);
    }
    const int rows = TBAT * THN * TSEQ;
    softmax_wei<<<dim3((rows + 255) / 256), dim3(256), 0, stream>>>(wei, rows, causal);
    // ctx[z] = wei_z (48x48) @ V_z (48x32); B = vt16 K-contiguous; K=48 tail
    {
      GemmP p = gp0();
      p.A = wei; p.lda = TSEQ; p.ziA = (long long)TSEQ * TSEQ;
      p.Bm = vt16; p.ldb = TSEQ;                  // B(k=s, n=d) = vt[..+ d*48 + s]
      p.zshB = 3; p.zmkB = 7; p.zoB = (long long)TSEQ * TE; p.ziB = (long long)THD * TSEQ;
      p.C = ctx; p.ldc = TE;
      p.zshC = 3; p.zmkC = 7; p.zoC = (long long)TSEQ * TE; p.ziC = THD;
      p.M = TSEQ; p.N = THD; p.K = TSEQ;
      launch_gemm(stream, p, TBAT * THN);
    }
    // output projection + bias + residual
    {
      GemmP p = gp0();
      p.A = ctx; p.lda = TE; p.Bm = wp_o; p.ldb = TE;
      p.bias = w.bo + (size_t)l * TE;
      p.R = resid; p.ldr = TE;
      p.C = out; p.ldc = TE;
      p.M = TMROW; p.N = TE; p.K = TE;
      launch_gemm(stream, p, 1);
    }
  };

  auto ffn = [&](const float* x, const FFNW& w, int l, const float* resid, float* out) {
    pack_rm(w.w1, (size_t)l * TE * TFF, wp_f1, TFF, TE);
    pack_rm(w.w2, (size_t)l * TFF * TE, wp_f2, TE, TFF);
    GemmP p1 = gp0();
    p1.A = x; p1.lda = TE; p1.Bm = wp_f1; p1.ldb = TE;
    p1.bias = w.b1 + (size_t)l * TFF;
    p1.C = ffh; p1.ldc = TFF; p1.M = TMROW; p1.N = TFF; p1.K = TE; p1.relu = 1;
    launch_gemm(stream, p1, 1);
    GemmP p2 = gp0();
    p2.A = ffh; p2.lda = TFF; p2.Bm = wp_f2; p2.ldb = TFF;
    p2.bias = w.b2 + (size_t)l * TE;
    p2.R = resid; p2.ldr = TE;
    p2.C = out; p2.ldc = TE; p2.M = TMROW; p2.N = TE; p2.K = TFF;
    launch_gemm(stream, p2, 1);
  };

  const unsigned embBlocks = (unsigned)(((long long)TMROW * TE + 255) / 256);

  // ---- Encoder ----
  embed_pe<<<dim3(embBlocks), dim3(256), 0, stream>>>(src, F(2), encx);
  for (int l = 0; l < TLAY; ++l) {
    mha(encx, encx, encx, enc_sa, l, /*causal=*/0, t1, encx);  // t1 = x + SA(x)
    layernorm(t1, enc_ln, l, t2);
    ffn(t2, enc_ff, l, t2, t1);                                // t1 = y + FFN(y)
    layernorm(t1, enc_ln, l, encx);
  }

  // ---- Decoder ----
  embed_pe<<<dim3(embBlocks), dim3(256), 0, stream>>>(trg, F(3), decx);
  for (int l = 0; l < TLAY; ++l) {
    mha(decx, decx, decx, dec_sa, l, /*causal=*/1, t1, decx);  // masked self-attn
    layernorm(t1, dec_ln, l, t2);                              // t2 = q
    mha(t2, encx, encx, dblk_sa, l, /*causal=*/1, t1, t2);     // cross-attn (ref applies trg_mask too)
    layernorm(t1, dblk_ln, l, decx);
    ffn(decx, dblk_ff, l, decx, t1);
    layernorm(t1, dblk_ln, l, decx);
  }

  // ---- Output head + softmax over sequence axis ----
  {
    pack_rm(out_w, 0, wp_out, TVOUT, TE);
    GemmP p = gp0();
    p.A = decx; p.lda = TE; p.Bm = wp_out; p.ldb = TE;
    p.bias = out_b;
    p.C = logits; p.ldc = TVOUT; p.M = TMROW; p.N = TVOUT; p.K = TE;
    launch_gemm(stream, p, 1);
  }
  softmax_seq<<<dim3((TBAT * TVOUT + 255) / 256), dim3(256), 0, stream>>>(
      logits, (float*)d_out);
}
